// AnomalyTransformer_83416854823357
// MI455X (gfx1250) — matrix operand order repaired and checked by reading, not compiled, not run on hardware
//
#include <hip/hip_runtime.h>
#include <hip/hip_bf16.h>
#include <math.h>
#include <stdint.h>

typedef __attribute__((ext_vector_type(16))) _Float16 v16h;
typedef __attribute__((ext_vector_type(8)))  _Float16 v8h;
typedef __attribute__((ext_vector_type(8)))  float    v8f;
typedef __attribute__((ext_vector_type(4)))  int      v4i;

// ---------------------------------------------------------------------------
// Async global->LDS staging support (CDNA5 GLOBAL_LOAD_ASYNC_TO_LDS_B128)
// ---------------------------------------------------------------------------
#if defined(__gfx1250__) && __has_builtin(__builtin_amdgcn_global_load_async_to_lds_b128)
  #define ASYNC_MODE 1
#elif defined(__gfx1250__)
  #define ASYNC_MODE 2
#else
  #define ASYNC_MODE 0
#endif

__device__ __forceinline__ void async_copy16(const void* gsrc, void* ldst) {
#if ASYNC_MODE == 1
  __builtin_amdgcn_global_load_async_to_lds_b128(
      (__attribute__((address_space(1))) v4i*)(uintptr_t)gsrc,
      (__attribute__((address_space(3))) v4i*)(uint32_t)(uintptr_t)ldst,
      0, 0);
#elif ASYNC_MODE == 2
  asm volatile("global_load_async_to_lds_b128 %0, %1, off"
               :: "v"((uint32_t)(uintptr_t)ldst), "v"((uint64_t)(uintptr_t)gsrc)
               : "memory");
#else
  *(uint4*)ldst = *(const uint4*)gsrc;
#endif
}

__device__ __forceinline__ void wait_async0() {
#if ASYNC_MODE == 1
#if __has_builtin(__builtin_amdgcn_s_wait_asynccnt)
  __builtin_amdgcn_s_wait_asynccnt(0);
#else
  asm volatile("s_wait_asynccnt 0" ::: "memory");
#endif
#elif ASYNC_MODE == 2
  asm volatile("s_wait_asynccnt 0" ::: "memory");
#endif
}

// ---------------------------------------------------------------------------
// WMMA helpers (gfx1250: D = A(16x32,f16) * B(32x16,f16) + C(16x16,f32))
// ---------------------------------------------------------------------------
__device__ __forceinline__ v8f wmma_f16(v16h a, v16h b, v8f c) {
  return __builtin_amdgcn_wmma_f32_16x16x32_f16(
      /*neg_a=*/false, a, /*neg_b=*/false, b,
      /*c_mod=*/(short)0, c, /*reuse_a=*/false, /*reuse_b=*/false);
}

// A fragment: 16x32 f16.  lane<16: M=lane, elems0..7=K[kh*8..], elems8..15=K[16+kh*8..]
__device__ __forceinline__ v16h frag_a_lds(const _Float16* base, int stride) {
  const int lane = threadIdx.x & 31;
  const int m = lane & 15;
  const int kh = lane >> 4;
  const _Float16* p = base + m * stride + kh * 8;
  v8h lo = *(const v8h*)(p);
  v8h hi = *(const v8h*)(p + 16);
  v16h r;
#pragma unroll
  for (int i = 0; i < 8; ++i) { r[i] = lo[i]; r[i + 8] = hi[i]; }
  return r;
}

// B fragment: 32x16 f16 from [N][K]-row-major LDS tile. lane&15 = column N,
// elems0..7=K[kg*8..], elems8..15=K[16+kg*8..].
__device__ __forceinline__ v16h frag_b_lds(const _Float16* base, int stride) {
  const int lane = threadIdx.x & 31;
  const int n = lane & 15;
  const int kg = lane >> 4;
  const _Float16* p = base + n * stride + kg * 8;
  v8h lo = *(const v8h*)(p);
  v8h hi = *(const v8h*)(p + 16);
  v16h r;
#pragma unroll
  for (int i = 0; i < 8; ++i) { r[i] = lo[i]; r[i + 8] = hi[i]; }
  return r;
}

// ---------------------------------------------------------------------------
// Generic WMMA GEMM:  Y[M,N] = act( A[M,K](f16) @ W[N,K]^T(f16) + bias[N] )
// block tile 128x128, 8 waves, wave tile 32x64 (2x4 wmma tiles), K step 32.
// Staging uses async global->LDS (ASYNCcnt) double buffering.
// ---------------------------------------------------------------------------
__global__ __launch_bounds__(256) void wmma_gemm_kernel(
    const _Float16* __restrict__ A, const _Float16* __restrict__ W,
    const float* __restrict__ bias, float* __restrict__ out32,
    _Float16* __restrict__ out16, int M, int N, int K, int act)
{
  constexpr int BKP = 32 + 8;                  // padded K stride (16B-aligned)
  __shared__ _Float16 sA[2][128][BKP];
  __shared__ _Float16 sB[2][128][BKP];

  const int t = threadIdx.x;
  const int row0 = blockIdx.y * 128;
  const int col0 = blockIdx.x * 128;
  const int lr = t >> 1;                        // staged row 0..127
  const int lc = (t & 1) * 16;                  // staged col 0 or 16
  const int nB = col0 + lr;                     // staged W row (bounds-checked)

  auto stage = [&](int buf, int k0) {
    {
      const _Float16* src = A + (size_t)(row0 + lr) * K + k0 + lc;
      async_copy16(src,     &sA[buf][lr][lc]);
      async_copy16(src + 8, &sA[buf][lr][lc + 8]);
    }
    if (nB < N) {
      const _Float16* src = W + (size_t)nB * K + k0 + lc;
      async_copy16(src,     &sB[buf][lr][lc]);
      async_copy16(src + 8, &sB[buf][lr][lc + 8]);
    } else {
      uint4 z = {0u, 0u, 0u, 0u};
      *(uint4*)&sB[buf][lr][lc]     = z;
      *(uint4*)&sB[buf][lr][lc + 8] = z;
    }
  };

  const int wid = t >> 5;
  const int wm = (wid >> 1) * 32;              // wave M offset 0/32/64/96
  const int wn = (wid & 1) * 64;               // wave N offset 0/64

  v8f acc[2][4];
  v8f zero = {0.f, 0.f, 0.f, 0.f, 0.f, 0.f, 0.f, 0.f};
#pragma unroll
  for (int i = 0; i < 2; ++i)
#pragma unroll
    for (int j = 0; j < 4; ++j) acc[i][j] = zero;

  stage(0, 0);
  wait_async0();
  __syncthreads();
  const int nk = K >> 5;
  for (int kb = 0; kb < nk; ++kb) {
    const int buf = kb & 1;
    if (kb + 1 < nk) stage(buf ^ 1, (kb + 1) << 5);   // async prefetch next
    v16h af[2], bf[4];
#pragma unroll
    for (int i = 0; i < 2; ++i) af[i] = frag_a_lds(&sA[buf][wm + i * 16][0], BKP);
#pragma unroll
    for (int j = 0; j < 4; ++j) bf[j] = frag_b_lds(&sB[buf][wn + j * 16][0], BKP);
#pragma unroll
    for (int i = 0; i < 2; ++i)
#pragma unroll
      for (int j = 0; j < 4; ++j) acc[i][j] = wmma_f16(af[i], bf[j], acc[i][j]);
    wait_async0();                                    // next-buffer loads done
    __syncthreads();
  }

  const int lane = t & 31;
  const int ccol = lane & 15;
  const int rsub = (lane >> 4) * 8;
#pragma unroll
  for (int i = 0; i < 2; ++i) {
#pragma unroll
    for (int j = 0; j < 4; ++j) {
      const int gc = col0 + wn + j * 16 + ccol;
      if (gc >= N) continue;
      const float bv = bias ? bias[gc] : 0.0f;
#pragma unroll
      for (int jj = 0; jj < 8; ++jj) {
        const int gr = row0 + wm + i * 16 + rsub + jj;
        float v = acc[i][j][jj] + bv;
        if (act) v = 0.5f * v * (1.0f + erff(v * 0.70710678f));   // exact GELU
        const size_t o = (size_t)gr * N + gc;
        if (out32) out32[o] = v;
        if (out16) out16[o] = (_Float16)v;
      }
    }
  }
}

// ---------------------------------------------------------------------------
// Fused masked attention per (b,h): S=QK^T/8 -> mask+softmax -> O=P@V (WMMA)
// L=100 padded to 112 (7 tiles of 16); P K-dim padded to 128.
// ---------------------------------------------------------------------------
__global__ __launch_bounds__(256) void attn_kernel(
    const _Float16* __restrict__ q16, const _Float16* __restrict__ k16,
    const _Float16* __restrict__ v16p, const unsigned char* __restrict__ mask,
    _Float16* __restrict__ o16)
{
  extern __shared__ char smem[];
  _Float16* Qs = (_Float16*)smem;              // [112][72]
  _Float16* Ks = Qs + 112 * 72;                // [112][72]
  _Float16* Vt = Ks + 112 * 72;                // [64][136]  (V transposed: [d][m])
  float*    S  = (float*)(Vt + 64 * 136);      // [112][112]
  _Float16* P  = (_Float16*)(S + 112 * 112);   // [112][128]

  const int t = threadIdx.x;
  const int bb = blockIdx.x >> 3;
  const int hh = blockIdx.x & 7;

  // stage Q, K (rows >= 100 zero-filled)
  for (int idx = t; idx < 112 * 64; idx += 256) {
    const int l = idx >> 6, d = idx & 63;
    _Float16 qv = (_Float16)0, kv = (_Float16)0;
    if (l < 100) {
      const size_t g = ((size_t)(bb * 100 + l)) * 512 + hh * 64 + d;
      qv = q16[g]; kv = k16[g];
    }
    Qs[l * 72 + d] = qv;
    Ks[l * 72 + d] = kv;
  }
  // stage V transposed with K padded to 128
  for (int idx = t; idx < 64 * 128; idx += 256) {
    const int d = idx >> 7, m = idx & 127;
    Vt[d * 136 + m] = (m < 100)
        ? v16p[((size_t)(bb * 100 + m)) * 512 + hh * 64 + d] : (_Float16)0;
  }
  __syncthreads();

  const int wid = t >> 5;
  const int lane = t & 31;
  const int ccol = lane & 15;
  const int rsub = (lane >> 4) * 8;
  v8f zero = {0.f, 0.f, 0.f, 0.f, 0.f, 0.f, 0.f, 0.f};

  // S = Q @ K^T * 0.125   (waves 0..6, one 16-row strip each)
  if (wid < 7) {
    const int rt = wid * 16;
    for (int nt = 0; nt < 7; ++nt) {
      v8f acc = zero;
#pragma unroll
      for (int k0 = 0; k0 < 64; k0 += 32) {
        v16h a = frag_a_lds(&Qs[rt * 72 + k0], 72);
        v16h b = frag_b_lds(&Ks[nt * 16 * 72 + k0], 72);
        acc = wmma_f16(a, b, acc);
      }
#pragma unroll
      for (int jj = 0; jj < 8; ++jj)
        S[(rt + rsub + jj) * 112 + nt * 16 + ccol] = acc[jj] * 0.125f;
    }
  }
  __syncthreads();

  // masked softmax over m, write P (f16), rows 0..99
  if (t < 100) {
    const int l = t;
    float mx = -1.0e30f;
    for (int m = 0; m < 100; ++m)
      if (mask[l * 100 + m]) mx = fmaxf(mx, S[l * 112 + m]);
    float sum = 0.0f;
    for (int m = 0; m < 100; ++m) {
      const float e = mask[l * 100 + m] ? __expf(S[l * 112 + m] - mx) : 0.0f;
      S[l * 112 + m] = e;
      sum += e;
    }
    const float inv = 1.0f / sum;
    for (int m = 0; m < 128; ++m)
      P[l * 128 + m] = (m < 100) ? (_Float16)(S[l * 112 + m] * inv) : (_Float16)0;
  }
  // zero pad rows 100..111 of P
  for (int idx = t; idx < 12 * 128; idx += 256)
    P[(100 + (idx >> 7)) * 128 + (idx & 127)] = (_Float16)0;
  __syncthreads();

  // O = P @ V   (M=112 strips, N=64, K=128)
  if (wid < 7) {
    const int rt = wid * 16;
    for (int nt = 0; nt < 4; ++nt) {
      v8f acc = zero;
#pragma unroll
      for (int k0 = 0; k0 < 128; k0 += 32) {
        v16h a = frag_a_lds(&P[rt * 128 + k0], 128);
        v16h b = frag_b_lds(&Vt[nt * 16 * 136 + k0], 136);
        acc = wmma_f16(a, b, acc);
      }
#pragma unroll
      for (int jj = 0; jj < 8; ++jj) {
        const int l = rt + rsub + jj;
        if (l < 100)
          o16[((size_t)(bb * 100 + l)) * 512 + hh * 64 + nt * 16 + ccol] =
              (_Float16)acc[jj];
      }
    }
  }
}

// ---------------------------------------------------------------------------
// LayerNorm over D=512; optional residual add; emits f32 + f16 copies
// ---------------------------------------------------------------------------
__global__ __launch_bounds__(256) void ln_kernel(
    const float* __restrict__ y, const float* __restrict__ res,
    const float* __restrict__ g, const float* __restrict__ bta,
    float* __restrict__ out32, _Float16* __restrict__ out16, int hasRes)
{
  __shared__ float s1[256];
  __shared__ float s2[256];
  const int row = blockIdx.x;
  const int t = threadIdx.x;
  const size_t base = (size_t)row * 512;
  float a = y[base + t], b = y[base + t + 256];
  if (hasRes) { a += res[base + t]; b += res[base + t + 256]; }
  s1[t] = a + b;
  s2[t] = a * a + b * b;
  __syncthreads();
  for (int s = 128; s > 0; s >>= 1) {
    if (t < s) { s1[t] += s1[t + s]; s2[t] += s2[t + s]; }
    __syncthreads();
  }
  const float mean = s1[0] * (1.0f / 512.0f);
  const float var  = s2[0] * (1.0f / 512.0f) - mean * mean;
  const float inv  = rsqrtf(var + 1e-5f);
  const float o0 = (a - mean) * inv * g[t] + bta[t];
  const float o1 = (b - mean) * inv * g[t + 256] + bta[t + 256];
  if (out32) { out32[base + t] = o0; out32[base + t + 256] = o1; }
  out16[base + t] = (_Float16)o0;
  out16[base + t + 256] = (_Float16)o1;
}

// ---------------------------------------------------------------------------
// Conv1d(k=3, circular) embedding + sinusoidal PE (computed on the fly)
// ---------------------------------------------------------------------------
__global__ __launch_bounds__(256) void embed_kernel(
    const float* __restrict__ x, const float* __restrict__ w,
    float* __restrict__ h32, _Float16* __restrict__ h16)
{
  const int bl = blockIdx.x;             // 0..25599
  const int b = bl / 100, l = bl % 100;
  const int t = threadIdx.x;
  __shared__ float xs[3][38];
  if (t < 114) {
    const int tap = t / 38, c = t % 38;
    const int ll = (l - 1 + tap + 100) % 100;
    xs[tap][c] = x[((size_t)b * 100 + ll) * 38 + c];
  }
  __syncthreads();
  for (int d = t; d < 512; d += 256) {
    float acc = 0.0f;
#pragma unroll
    for (int tap = 0; tap < 3; ++tap)
      for (int c = 0; c < 38; ++c)
        acc += xs[tap][c] * w[d * 114 + c * 3 + tap];
    const int i2 = d & ~1;
    const float freq = __expf(-(float)i2 * 0.0179889463f);   // ln(1e4)/512
    const float ang = (float)l * freq;
    acc += (d & 1) ? __cosf(ang) : __sinf(ang);
    h32[(size_t)bl * 512 + d] = acc;
    h16[(size_t)bl * 512 + d] = (_Float16)acc;
  }
}

__global__ void pack_f16_kernel(const float* __restrict__ src,
                                _Float16* __restrict__ dst, int n) {
  const int i = blockIdx.x * 256 + threadIdx.x;
  if (i < n) dst[i] = (_Float16)src[i];
}

// ---------------------------------------------------------------------------
extern "C" void kernel_launch(void* const* d_in, const int* in_sizes, int n_in,
                              void* d_out, int out_size, void* d_ws, size_t ws_size,
                              hipStream_t stream) {
  (void)in_sizes; (void)n_in; (void)out_size; (void)ws_size;
  const float* x      = (const float*)d_in[0];
  const float* emb_w  = (const float*)d_in[1];
  const float* Wq = (const float*)d_in[2];   const float* bq = (const float*)d_in[3];
  const float* Wk = (const float*)d_in[4];   const float* bk = (const float*)d_in[5];
  const float* Wv = (const float*)d_in[6];   const float* bv = (const float*)d_in[7];
  const float* Wo = (const float*)d_in[8];   const float* bo = (const float*)d_in[9];
  const float* c1w = (const float*)d_in[10]; const float* c1b = (const float*)d_in[11];
  const float* c2w = (const float*)d_in[12]; const float* c2b = (const float*)d_in[13];
  const float* ln1s = (const float*)d_in[14]; const float* ln1b = (const float*)d_in[15];
  const float* ln2s = (const float*)d_in[16]; const float* ln2b = (const float*)d_in[17];
  const float* lnfs = (const float*)d_in[18]; const float* lnfb = (const float*)d_in[19];
  const float* proj_w = (const float*)d_in[20]; const float* proj_b = (const float*)d_in[21];
  const unsigned char* mask = (const unsigned char*)d_in[22];

  char* ws = (char*)d_ws;
  size_t off = 0;
  auto take = [&](size_t bytes) {
    char* p = ws + off;
    off += (bytes + 255) & ~(size_t)255;
    return p;
  };
  constexpr size_t ML = 25600;                     // B*L rows
  float*    h32  = (float*)take(ML * 512 * 4);
  float*    y32  = (float*)take(ML * 512 * 4);
  _Float16* h16  = (_Float16*)take(ML * 512 * 2);
  _Float16* q16  = (_Float16*)take(ML * 512 * 2);  // reused as FFN-mid buffer
  _Float16* k16  = (_Float16*)take(ML * 512 * 2);
  _Float16* v16w = (_Float16*)take(ML * 512 * 2);
  _Float16* o16  = (_Float16*)take(ML * 512 * 2);
  _Float16* wq16 = (_Float16*)take((size_t)3 * 512 * 512 * 2);
  _Float16* wk16 = (_Float16*)take((size_t)3 * 512 * 512 * 2);
  _Float16* wv16 = (_Float16*)take((size_t)3 * 512 * 512 * 2);
  _Float16* wo16 = (_Float16*)take((size_t)3 * 512 * 512 * 2);
  _Float16* c1w16 = (_Float16*)take((size_t)3 * 512 * 512 * 2);
  _Float16* c2w16 = (_Float16*)take((size_t)3 * 512 * 512 * 2);
  _Float16* pw16 = (_Float16*)take((size_t)38 * 512 * 2);

  auto pack = [&](const float* s, _Float16* d, int n) {
    pack_f16_kernel<<<(n + 255) / 256, 256, 0, stream>>>(s, d, n);
  };
  const int WN = 3 * 512 * 512;
  pack(Wq, wq16, WN);  pack(Wk, wk16, WN);  pack(Wv, wv16, WN);  pack(Wo, wo16, WN);
  pack(c1w, c1w16, WN); pack(c2w, c2w16, WN);
  pack(proj_w, pw16, 38 * 512);

  embed_kernel<<<ML, 256, 0, stream>>>(x, emb_w, h32, h16);

  constexpr int ATTN_LDS = (112 * 72 + 112 * 72 + 64 * 136) * 2
                         + 112 * 112 * 4 + 112 * 128 * 2;   // 128512 B
  (void)hipFuncSetAttribute(reinterpret_cast<const void*>(&attn_kernel),
                            hipFuncAttributeMaxDynamicSharedMemorySize, ATTN_LDS);

  const dim3 gD(4, 200);   // N=512 -> 4 col-tiles, M=25600 -> 200 row-tiles
  for (int l = 0; l < 3; ++l) {
    const size_t wOff = (size_t)l * 512 * 512;
    const size_t bOff = (size_t)l * 512;
    wmma_gemm_kernel<<<gD, 256, 0, stream>>>(h16, wq16 + wOff, bq + bOff,
                                             nullptr, q16, 25600, 512, 512, 0);
    wmma_gemm_kernel<<<gD, 256, 0, stream>>>(h16, wk16 + wOff, bk + bOff,
                                             nullptr, k16, 25600, 512, 512, 0);
    wmma_gemm_kernel<<<gD, 256, 0, stream>>>(h16, wv16 + wOff, bv + bOff,
                                             nullptr, v16w, 25600, 512, 512, 0);
    attn_kernel<<<2048, 256, ATTN_LDS, stream>>>(q16, k16, v16w, mask, o16);
    wmma_gemm_kernel<<<gD, 256, 0, stream>>>(o16, wo16 + wOff, bo + bOff,
                                             y32, nullptr, 25600, 512, 512, 0);
    ln_kernel<<<ML, 256, 0, stream>>>(y32, h32, ln1s + bOff, ln1b + bOff,
                                      h32, h16, 1);
    wmma_gemm_kernel<<<gD, 256, 0, stream>>>(h16, c1w16 + wOff, c1b + bOff,
                                             nullptr, q16, 25600, 512, 512, 1);
    wmma_gemm_kernel<<<gD, 256, 0, stream>>>(q16, c2w16 + wOff, c2b + bOff,
                                             y32, nullptr, 25600, 512, 512, 0);
    ln_kernel<<<ML, 256, 0, stream>>>(y32, h32, ln2s + bOff, ln2b + bOff,
                                      h32, h16, 1);
  }
  ln_kernel<<<ML, 256, 0, stream>>>(h32, nullptr, lnfs, lnfb, nullptr, h16, 0);
  wmma_gemm_kernel<<<dim3(1, 200), 256, 0, stream>>>(h16, pw16, proj_b,
                                                     (float*)d_out, nullptr,
                                                     25600, 38, 512, 0);
}
